// Encoder_89026082111488
// MI455X (gfx1250) — matrix operand order repaired and checked by reading, not compiled, not run on hardware
//
#include <hip/hip_runtime.h>
#include <hip/hip_bf16.h>
#include <stdint.h>

typedef unsigned short     u16;
typedef unsigned int       u32;
typedef unsigned long long u64;

typedef __attribute__((ext_vector_type(16))) __bf16 v16bf;
typedef __attribute__((ext_vector_type(8)))  float  v8f;

constexpr int Bd = 64;     // batch
constexpr int Td = 256;    // time
constexpr int Ed = 512;    // embed
constexpr int Ud = 1024;   // units
constexpr int Gd = 4096;   // 4*U (gate-major columns: i,f,c,o)

// ---------- bf16 helpers (RNE) ----------
__device__ __forceinline__ u16 f2bf(float f) {
  u32 u = __float_as_uint(f);
  u = (u + 0x7fffu + ((u >> 16) & 1u)) >> 16;
  return (u16)u;
}
__device__ __forceinline__ float bf2f(u16 s) {
  return __uint_as_float(((u32)s) << 16);
}

union FragBF { v16bf v; u16 u[16]; u32 d[8]; };

// ---------- CDNA5 async global->LDS copies (ASYNCcnt-tracked) ----------
__device__ __forceinline__ u32 lds_off(const void* p) {
  // flat LDS aperture: low 32 bits are the LDS byte offset
  return (u32)(uintptr_t)p;
}
__device__ __forceinline__ void a_ld_b128(u32 lds, const void* g) {
  asm volatile("global_load_async_to_lds_b128 %0, %1, off"
               :: "v"(lds), "v"((u64)(uintptr_t)g) : "memory");
}
__device__ __forceinline__ void a_ld_b64(u32 lds, const void* g) {
  asm volatile("global_load_async_to_lds_b64 %0, %1, off"
               :: "v"(lds), "v"((u64)(uintptr_t)g) : "memory");
}

// ---------- utility kernels ----------
__global__ void k_cvt_bf16(const float* __restrict__ src, u16* __restrict__ dst, int n) {
  int i = blockIdx.x * 256 + threadIdx.x;
  if (i < n) dst[i] = f2bf(src[i]);
}
// transpose-convert: src f32 [K][4096] -> dst bf16 [4096][K]; K = 1<<logK
__global__ void k_cvt_t_bf16(const float* __restrict__ src, u16* __restrict__ dst,
                             int logK, int n) {
  int i = blockIdx.x * 256 + threadIdx.x;
  if (i < n) {
    int col = i >> logK;            // n in [0,4096)
    int k   = i & ((1 << logK) - 1);
    dst[i] = f2bf(src[(size_t)k * Gd + col]);
  }
}
__global__ void k_zero_u32(u32* __restrict__ p, int n) {
  int i = blockIdx.x * 256 + threadIdx.x;
  if (i < n) p[i] = 0u;
}
__global__ void k_copy_f32(const float* __restrict__ s, float* __restrict__ d, int n) {
  int i = blockIdx.x * 256 + threadIdx.x;
  if (i < n) d[i] = s[i];
}

// ---------------------------------------------------------------------------
// Input projection: xw[t,b,:] = A[b*T+t,:] @ W + bias
//   A  bf16 [B*T][K] row-major, Wt bf16 [4096][K] (pre-transposed)
// Output TIME-MAJOR bf16 [T][B][4096].
// Block tile 128(M)x64(N), 8 waves (4x2), 2x2 WMMA accumulators per wave.
// Double-buffered LDS filled by async global->LDS copies (3 per wave per chunk).
// ---------------------------------------------------------------------------
__global__ __launch_bounds__(256)
void k_gemm_xw(const u16* __restrict__ A, const u16* __restrict__ Wt,
               const float* __restrict__ bias, u16* __restrict__ xw, int K) {
  __shared__ u16 sA[2][128][40];   // [m][k], 80B rows (16B aligned)
  __shared__ u16 sB[2][64][40];    // [n][k]

  const int tid  = threadIdx.x;
  const int wid  = tid >> 5, lane = tid & 31;
  const int wm   = wid & 3;          // 4 waves over M (32 rows each)
  const int wn   = wid >> 2;         // 2 waves over N (32 cols each)
  const int hh   = lane >> 4;        // lane half
  const int cc   = lane & 15;
  const int n0   = blockIdx.x * 64;
  const int m0   = blockIdx.y * 128;

  auto pref = [&](int buf, int k0) {
#pragma unroll
    for (int r = 0; r < 2; ++r) {                 // A tile: 128 rows x 4 b128 chunks
      int idx = tid + r * 256;
      int row = idx >> 2, ch = idx & 3;
      a_ld_b128(lds_off(&sA[buf][row][ch * 8]),
                A + (size_t)(m0 + row) * K + k0 + ch * 8);
    }
    {                                             // B tile: 64 rows x 4 b128 chunks
      int n = tid >> 2, ch = tid & 3;
      a_ld_b128(lds_off(&sB[buf][n][ch * 8]),
                Wt + (size_t)(n0 + n) * K + k0 + ch * 8);
    }
  };

  v8f acc[2][2];
  const v8f vzero = {0.f,0.f,0.f,0.f,0.f,0.f,0.f,0.f};
  acc[0][0] = vzero; acc[0][1] = vzero; acc[1][0] = vzero; acc[1][1] = vzero;

  const int nIter = K >> 5;
  pref(0, 0);
  for (int it = 0; it < nIter; ++it) {
    const int cur = it & 1;
    if (it + 1 < nIter) {
      pref(cur ^ 1, (it + 1) * 32);
      asm volatile("s_wait_asynccnt 0x3" ::: "memory");  // 3 just issued -> cur resident
    } else {
      asm volatile("s_wait_asynccnt 0x0" ::: "memory");
    }
    __syncthreads();

    FragBF fa[2], fb[2];
#pragma unroll
    for (int i = 0; i < 2; ++i) {
      const u16* pr = &sA[cur][wm * 32 + i * 16 + cc][0];
#pragma unroll
      for (int q = 0; q < 4; ++q) fa[i].d[q]     = *(const u32*)(pr + hh * 8 + 2 * q);
#pragma unroll
      for (int q = 0; q < 4; ++q) fa[i].d[4 + q] = *(const u32*)(pr + 16 + hh * 8 + 2 * q);
    }
#pragma unroll
    for (int j = 0; j < 2; ++j) {
      const u16* pc = &sB[cur][wn * 32 + j * 16 + cc][0];
#pragma unroll
      for (int q = 0; q < 8; ++q) fb[j].d[q] = *(const u32*)(pc + (q >> 2) * 16 + hh * 8 + 2 * (q & 3));
    }
    __syncthreads();   // all frag reads done before this buffer is refilled

#pragma unroll
    for (int i = 0; i < 2; ++i)
#pragma unroll
      for (int j = 0; j < 2; ++j)
        acc[i][j] = __builtin_amdgcn_wmma_f32_16x16x32_bf16(
            false, fa[i].v, false, fb[j].v, (short)0, acc[i][j], false, false);
  }

  // epilogue: add bias, remap (b*T+t) -> time-major [t][b], store bf16
#pragma unroll
  for (int i = 0; i < 2; ++i)
#pragma unroll
    for (int j = 0; j < 2; ++j) {
      int gn = n0 + wn * 32 + j * 16 + cc;
      float bv = bias[gn];
#pragma unroll
      for (int e = 0; e < 8; ++e) {
        int gm = m0 + wm * 32 + i * 16 + e + hh * 8;
        int bb = gm / Td, tt = gm % Td;   // Td=256 -> shifts
        float v = acc[i][j][e] + bv;
        xw[((size_t)tt * Bd + bb) * Gd + gn] = f2bf(v);
      }
    }
}

// ---------------------------------------------------------------------------
// Recurrent step GEMM: z[64][4096] = xw_t + (h*mask[gate]) @ U_rec
//   gate = column/1024; accumulators seeded from xw_t (C operand of WMMA chain)
//   Urt bf16 [4096][1024] pre-transposed. Block tile 64x32, 8 waves (4x2),
//   1 accumulator per wave, 128 WGs, async double-buffered K pipeline.
// ---------------------------------------------------------------------------
__global__ __launch_bounds__(256)
void k_rec_gemm(const u16* __restrict__ xwt,   // bf16 [64][4096] at time t
                const u16* __restrict__ hm,    // bf16 [4][64][1024]  (h * mask[g])
                const u16* __restrict__ Urt,   // bf16 [4096][1024] transposed
                float* __restrict__ z) {       // f32 [64][4096]
  __shared__ u16 sA[2][64][40];
  __shared__ u16 sB[2][32][40];

  const int tid  = threadIdx.x;
  const int wid  = tid >> 5, lane = tid & 31;
  const int wm   = wid & 3;          // 4 waves over M (16 rows each)
  const int wn   = wid >> 2;         // 2 waves over N (16 cols each)
  const int hh   = lane >> 4;
  const int cc   = lane & 15;
  const int n0   = blockIdx.x * 32;
  const int gate = n0 >> 10;
  const u16* Ag  = hm + (size_t)gate * Bd * Ud;

  auto pref = [&](int buf, int k0) {
    {                                 // A tile: 64 rows x 4 b128 chunks, 1/thread
      int row = tid >> 2, ch = tid & 3;
      a_ld_b128(lds_off(&sA[buf][row][ch * 8]),
                Ag + (size_t)row * Ud + k0 + ch * 8);
    }
    {                                 // B tile: 32 rows x 8 b64 chunks, 1/thread
      int n = tid >> 3, ch = tid & 7;
      a_ld_b64(lds_off(&sB[buf][n][ch * 4]),
               Urt + (size_t)(n0 + n) * Ud + k0 + ch * 4);
    }
  };

  const int gn = n0 + wn * 16 + cc;

  // seed accumulator with the precomputed input projection
  v8f acc;
#pragma unroll
  for (int e = 0; e < 8; ++e) {
    int row = wm * 16 + e + hh * 8;
    acc[e] = bf2f(xwt[(size_t)row * Gd + gn]);
  }

  const int nIter = Ud >> 5;   // 32
  pref(0, 0);
  for (int it = 0; it < nIter; ++it) {
    const int cur = it & 1;
    if (it + 1 < nIter) {
      pref(cur ^ 1, (it + 1) * 32);
      asm volatile("s_wait_asynccnt 0x2" ::: "memory");  // 2 just issued per wave
    } else {
      asm volatile("s_wait_asynccnt 0x0" ::: "memory");
    }
    __syncthreads();

    FragBF fa, fb;
    const u16* pr = &sA[cur][wm * 16 + cc][0];
#pragma unroll
    for (int q = 0; q < 4; ++q) fa.d[q]     = *(const u32*)(pr + hh * 8 + 2 * q);
#pragma unroll
    for (int q = 0; q < 4; ++q) fa.d[4 + q] = *(const u32*)(pr + 16 + hh * 8 + 2 * q);
    const u16* pc = &sB[cur][wn * 16 + cc][0];
#pragma unroll
    for (int q = 0; q < 8; ++q) fb.d[q] = *(const u32*)(pc + (q >> 2) * 16 + hh * 8 + 2 * (q & 3));
    __syncthreads();

    acc = __builtin_amdgcn_wmma_f32_16x16x32_bf16(
        false, fa.v, false, fb.v, (short)0, acc, false, false);
  }

#pragma unroll
  for (int e = 0; e < 8; ++e) {
    int row = wm * 16 + e + hh * 8;
    z[(size_t)row * Gd + gn] = acc[e];
  }
}

// ---------------------------------------------------------------------------
// Gate nonlinearities + state update + masked-h (bf16) for next step.
// ---------------------------------------------------------------------------
__global__ __launch_bounds__(256)
void k_gates(const float* __restrict__ z, const float* __restrict__ mask,
             float* __restrict__ h, float* __restrict__ c,
             float* __restrict__ outh, u16* __restrict__ hm, int t) {
  int idx = blockIdx.x * 256 + threadIdx.x;      // 0 .. B*U-1
  int b = idx >> 10, u = idx & 1023;
  const float* zr = z + (size_t)b * Gd;
  float zi = zr[u];
  float zf = zr[Ud + u];
  float zc = zr[2 * Ud + u];
  float zo = zr[3 * Ud + u];

  float ig = 1.f / (1.f + __expf(-zi));
  float fg = 1.f / (1.f + __expf(-zf));
  float gg = tanhf(zc);
  float og = 1.f / (1.f + __expf(-zo));

  float cn = fg * c[idx] + ig * gg;
  float hn = og * tanhf(cn);
  c[idx] = cn;
  h[idx] = hn;
  outh[((size_t)b * Td + t) * Ud + u] = hn;
#pragma unroll
  for (int g = 0; g < 4; ++g)
    hm[(size_t)g * Bd * Ud + idx] = f2bf(hn * mask[(size_t)g * Bd * Ud + idx]);
}

// ---------------------------------------------------------------------------
extern "C" void kernel_launch(void* const* d_in, const int* in_sizes, int n_in,
                              void* d_out, int out_size, void* d_ws, size_t ws_size,
                              hipStream_t stream) {
  (void)in_sizes; (void)n_in; (void)out_size; (void)ws_size;
  const float* x  = (const float*)d_in[0];
  const float* W1 = (const float*)d_in[1];
  const float* U1 = (const float*)d_in[2];
  const float* b1 = (const float*)d_in[3];
  const float* W2 = (const float*)d_in[4];
  const float* U2 = (const float*)d_in[5];
  const float* b2 = (const float*)d_in[6];
  const float* m1 = (const float*)d_in[7];
  const float* m2 = (const float*)d_in[8];
  float* out = (float*)d_out;

  char* ws = (char*)d_ws;
  size_t off = 0;
  auto take = [&](size_t bytes) -> char* {
    char* p = ws + off;
    off = (off + bytes + 255) & ~(size_t)255;
    return p;
  };
  u16*   W1t  = (u16*)take((size_t)Ed * Gd * 2);          //  4 MB  [4096][512]
  u16*   U1t  = (u16*)take((size_t)Ud * Gd * 2);          //  8 MB  [4096][1024]
  u16*   W2t  = (u16*)take((size_t)Ud * Gd * 2);          //  8 MB  [4096][1024]
  u16*   U2t  = (u16*)take((size_t)Ud * Gd * 2);          //  8 MB  [4096][1024]
  u16*   xb   = (u16*)take((size_t)Bd * Td * Ud * 2);     // 32 MB (layer input bf16)
  u16*   xw   = (u16*)take((size_t)Td * Bd * Gd * 2);     // 128 MB (time-major preact)
  float* out1 = (float*)take((size_t)Bd * Td * Ud * 4);   // 64 MB
  float* zbuf = (float*)take((size_t)Bd * Gd * 4);        //  1 MB
  float* hbuf = (float*)take((size_t)Bd * Ud * 4);        // h,c,hm contiguous for zeroing
  float* cbuf = (float*)take((size_t)Bd * Ud * 4);
  u16*   hm   = (u16*)take((size_t)4 * Bd * Ud * 2);

  auto cvt = [&](const float* s, u16* d, int n) {
    k_cvt_bf16<<<(n + 255) / 256, 256, 0, stream>>>(s, d, n);
  };
  auto cvtT = [&](const float* s, u16* d, int logK) {
    int n = Gd << logK;
    k_cvt_t_bf16<<<(n + 255) / 256, 256, 0, stream>>>(s, d, logK, n);
  };

  // one-time transposed bf16 weight conversions ([K][4096] -> [4096][K])
  cvtT(W1, W1t, 9);    // K = 512
  cvtT(U1, U1t, 10);   // K = 1024
  cvtT(W2, W2t, 10);
  cvtT(U2, U2t, 10);

  const dim3 gridXW(Gd / 64, (Bd * Td) / 128);
  const int nStateU32 = (int)(((size_t)Bd * Ud * 4 * 2 + (size_t)4 * Bd * Ud * 2) / 4);

  // ---------------- layer 1 ----------------
  cvt(x, xb, Bd * Td * Ed);
  k_gemm_xw<<<gridXW, 256, 0, stream>>>(xb, W1t, b1, xw, Ed);
  k_zero_u32<<<(nStateU32 + 255) / 256, 256, 0, stream>>>((u32*)hbuf, nStateU32);
  for (int t = 0; t < Td; ++t) {
    k_rec_gemm<<<Gd / 32, 256, 0, stream>>>(xw + (size_t)t * Bd * Gd, hm, U1t, zbuf);
    k_gates<<<(Bd * Ud) / 256, 256, 0, stream>>>(zbuf, m1, hbuf, cbuf, out1, hm, t);
  }

  // ---------------- layer 2 ----------------
  cvt(out1, xb, Bd * Td * Ud);
  k_gemm_xw<<<gridXW, 256, 0, stream>>>(xb, W2t, b2, xw, Ud);
  k_zero_u32<<<(nStateU32 + 255) / 256, 256, 0, stream>>>((u32*)hbuf, nStateU32);
  for (int t = 0; t < Td; ++t) {
    k_rec_gemm<<<Gd / 32, 256, 0, stream>>>(xw + (size_t)t * Bd * Gd, hm, U2t, zbuf);
    k_gates<<<(Bd * Ud) / 256, 256, 0, stream>>>(zbuf, m2, hbuf, cbuf, out, hm, t);
  }

  // final h2, c2 appended after out2 [B,T,U]
  k_copy_f32<<<((Bd * Ud) + 255) / 256, 256, 0, stream>>>(hbuf, out + (size_t)Bd * Td * Ud, Bd * Ud);
  k_copy_f32<<<((Bd * Ud) + 255) / 256, 256, 0, stream>>>(cbuf, out + (size_t)Bd * Td * Ud + Bd * Ud, Bd * Ud);
}
